// AttentionBlock_69836168232991
// MI455X (gfx1250) — matrix operand order repaired and checked by reading, not compiled, not run on hardware
//
#include <hip/hip_runtime.h>
#include <hip/hip_bf16.h>

typedef __attribute__((ext_vector_type(16))) _Float16 v16h;
typedef __attribute__((ext_vector_type(8)))  _Float16 v8h;
typedef __attribute__((ext_vector_type(8)))  float    v8f;

#define BATCH  16
#define CHAN   512
#define HW     1024
#define HEADS  8
#define HDIM   64
#define GROUPS 32
#define CPG    16
#define OQKV   1536
#define TT     32   // attention query-tile rows per workgroup
#define SLD    (HW + 16)  // f16 LDS row stride (2080 B -> +8 bank rotation)

// ---------------- WMMA fragment helpers (v_wmma_f32_16x16x32_f16) ----------------
// A (16x32, MxK): lane l holds row M=l&15; lane-half (l>>4) selects K-subgroup.
//   element j (0..15) -> K = (l>>4)*16 + j
__device__ __forceinline__ v16h frag_a_rm(const _Float16* __restrict__ A, int ldA) {
  const int lane = threadIdx.x & 31;
  const int row  = lane & 15;
  const int hi   = lane >> 4;
  const v8h p0 = *(const v8h*)(A + (size_t)row * ldA + hi * 16);
  const v8h p1 = *(const v8h*)(A + (size_t)row * ldA + hi * 16 + 8);
  v16h a;
#pragma unroll
  for (int j = 0; j < 8; ++j) { a[j] = p0[j]; a[8 + j] = p1[j]; }
  return a;
}

// B (32x16, KxN): lane l holds col N=l&15; element j -> K=(l>>4)*16+j.
// Memory layout is [n][k] row-major (k contiguous) -> two b128 loads per lane.
__device__ __forceinline__ v16h frag_b_kc(const _Float16* __restrict__ Bt, int ldB) {
  const int lane = threadIdx.x & 31;
  const int col  = lane & 15;
  const int kg   = lane >> 4;
  const v8h p0 = *(const v8h*)(Bt + (size_t)col * ldB + kg * 16);
  const v8h p1 = *(const v8h*)(Bt + (size_t)col * ldB + kg * 16 + 8);
  v16h b;
#pragma unroll
  for (int j = 0; j < 8; ++j) { b[j] = p0[j]; b[8 + j] = p1[j]; }
  return b;
}

__device__ __forceinline__ v8f wmma32(v16h a, v16h b, v8f c) {
  return __builtin_amdgcn_wmma_f32_16x16x32_f16(false, a, false, b, (short)0, c,
                                                false, false);
}

// ---------------- 1) weight fp32 -> f16 ----------------
__global__ __launch_bounds__(256) void k_wcvt(const float* __restrict__ qkv_w,
                                              const float* __restrict__ proj_w,
                                              _Float16* __restrict__ wq,
                                              _Float16* __restrict__ wp) {
  int i = blockIdx.x * 256 + threadIdx.x;
  if (i < OQKV * CHAN) wq[i] = (_Float16)qkv_w[i];
  if (i < CHAN * CHAN) wp[i] = (_Float16)proj_w[i];
}

// ---------------- 2) GroupNorm -> xnT [b, hw, c] f16 ----------------
__global__ __launch_bounds__(256) void k_gn(const float* __restrict__ x,
                                            const float* __restrict__ gamma,
                                            const float* __restrict__ beta,
                                            _Float16* __restrict__ xnT) {
  __shared__ float rs[256], rss[256];
  const int b  = blockIdx.x >> 5;
  const int g  = blockIdx.x & 31;
  const int c0 = g * CPG;
  const float* xg = x + ((size_t)b * CHAN + c0) * HW;
  const int tid = threadIdx.x;
  float s = 0.f, ss = 0.f;
  for (int i = tid; i < CPG * HW; i += 256) {
    float v = xg[i];
    s += v; ss += v * v;
  }
  rs[tid] = s; rss[tid] = ss;
  __syncthreads();
  for (int w = 128; w > 0; w >>= 1) {
    if (tid < w) { rs[tid] += rs[tid + w]; rss[tid] += rss[tid + w]; }
    __syncthreads();
  }
  const float invN = 1.f / (float)(CPG * HW);
  const float mean = rs[0] * invN;
  const float var  = rss[0] * invN - mean * mean;
  const float inv  = rsqrtf(var + 1e-5f);
  for (int i = tid; i < CPG * HW; i += 256) {
    int c = i >> 10, t = i & (HW - 1);
    int ch = c0 + c;
    float v = (xg[i] - mean) * inv * gamma[ch] + beta[ch];
    xnT[((size_t)b * HW + t) * CHAN + ch] = (_Float16)v;
  }
}

// ---------------- 3) QKV GEMM: [1536,512] x [512,1024] per batch ----------------
// q,k -> [b, H, hw, 64]; v -> [b, H, 64, hw] (transposed so PV B-frags vectorize)
__global__ __launch_bounds__(256) void k_gemm_qkv(const _Float16* __restrict__ wq,
                                                  const _Float16* __restrict__ xnT,
                                                  const float* __restrict__ qkv_b,
                                                  _Float16* __restrict__ qb,
                                                  _Float16* __restrict__ kb,
                                                  _Float16* __restrict__ vb) {
  const int b  = blockIdx.z;
  const int m0 = blockIdx.y * 128;
  const int n0 = blockIdx.x * 128;
  const int w  = threadIdx.x >> 5;
  const int mBase = m0 + (w >> 1) * 32;
  const int nBase = n0 + (w & 1) * 64;
  const _Float16* Bbase = xnT + (size_t)b * HW * CHAN;

  const v8f zv = {0.f, 0.f, 0.f, 0.f, 0.f, 0.f, 0.f, 0.f};
  v8f acc[2][4];
#pragma unroll
  for (int i = 0; i < 2; ++i)
#pragma unroll
    for (int j = 0; j < 4; ++j) acc[i][j] = zv;

  for (int k0 = 0; k0 < CHAN; k0 += 32) {
    v16h a0 = frag_a_rm(wq + (size_t)mBase * CHAN + k0, CHAN);
    v16h a1 = frag_a_rm(wq + (size_t)(mBase + 16) * CHAN + k0, CHAN);
#pragma unroll
    for (int nt = 0; nt < 4; ++nt) {
      v16h bf = frag_b_kc(Bbase + (size_t)(nBase + nt * 16) * CHAN + k0, CHAN);
      acc[0][nt] = wmma32(a0, bf, acc[0][nt]);
      acc[1][nt] = wmma32(a1, bf, acc[1][nt]);
    }
  }

  const int lane = threadIdx.x & 31;
  const int cm = (lane >> 4) * 8;
  const int cn = lane & 15;
#pragma unroll
  for (int mt = 0; mt < 2; ++mt)
#pragma unroll
    for (int nt = 0; nt < 4; ++nt)
#pragma unroll
      for (int i = 0; i < 8; ++i) {
        int o = mBase + mt * 16 + cm + i;
        int t = nBase + nt * 16 + cn;
        float v = acc[mt][nt][i] + qkv_b[o];
        if (o < CHAN) {
          int hh = o >> 6, dd = o & 63;
          qb[(((size_t)b * HEADS + hh) * HW + t) * HDIM + dd] = (_Float16)v;
        } else if (o < 2 * CHAN) {
          int oo = o - CHAN, hh = oo >> 6, dd = oo & 63;
          kb[(((size_t)b * HEADS + hh) * HW + t) * HDIM + dd] = (_Float16)v;
        } else {
          int oo = o - 2 * CHAN, hh = oo >> 6, dd = oo & 63;
          vb[(((size_t)b * HEADS + hh) * HDIM + dd) * HW + t] = (_Float16)v;
        }
      }
}

// ---------------- 4) attention: score/P slab lives in LDS as f16 ----------------
__global__ __launch_bounds__(256) void k_attn(const _Float16* __restrict__ qb,
                                              const _Float16* __restrict__ kb,
                                              const _Float16* __restrict__ vb,
                                              _Float16* __restrict__ ob) {
  __shared__ _Float16 Sh[TT][SLD];  // 32 x 1040 f16 = 65 KB
  __shared__ float red[TT][8];

  const int bid = blockIdx.x;
  const int b  = bid >> 8;
  const int h  = (bid >> 5) & 7;
  const int t0 = (bid & 31) * TT;
  const size_t headOff = ((size_t)b * HEADS + h) * (size_t)HW * HDIM;
  const _Float16* Q  = qb + headOff;
  const _Float16* K  = kb + headOff;
  const _Float16* Vt = vb + headOff;  // [d][s] layout, s contiguous

  const int w    = threadIdx.x >> 5;
  const int lane = threadIdx.x & 31;
  const int cm   = (lane >> 4) * 8;
  const int cn   = lane & 15;
  const v8f zv = {0.f, 0.f, 0.f, 0.f, 0.f, 0.f, 0.f, 0.f};

  // S = Q Kᵀ : 2 m-tiles x 64 n-tiles of 16x16; 8 waves, 16 n-tiles each
  {
    const int mt = w >> 2;
    const int nq = w & 3;
    v16h a0 = frag_a_rm(Q + (size_t)(t0 + mt * 16) * HDIM + 0, HDIM);
    v16h a1 = frag_a_rm(Q + (size_t)(t0 + mt * 16) * HDIM + 32, HDIM);
    for (int nt = nq * 16; nt < nq * 16 + 16; ++nt) {
      const int s0 = nt * 16;
      v16h b0 = frag_b_kc(K + (size_t)s0 * HDIM + 0, HDIM);
      v16h b1 = frag_b_kc(K + (size_t)s0 * HDIM + 32, HDIM);
      v8f c = zv;
      c = wmma32(a0, b0, c);
      c = wmma32(a1, b1, c);
#pragma unroll
      for (int i = 0; i < 8; ++i) Sh[mt * 16 + cm + i][s0 + cn] = (_Float16)c[i];
    }
  }
  __syncthreads();

  // exact softmax (scale = 1/sqrt(64) = 0.125), stats in f32, P stored as f16
  {
    const int r    = threadIdx.x >> 3;
    const int slot = threadIdx.x & 7;
    const int c0   = slot * 128;
    float mx = -3.0e38f;
    for (int c = c0; c < c0 + 128; ++c) mx = fmaxf(mx, (float)Sh[r][c]);
    red[r][slot] = mx;
    __syncthreads();
    mx = red[r][0];
#pragma unroll
    for (int i = 1; i < 8; ++i) mx = fmaxf(mx, red[r][i]);
    float sum = 0.f;
    for (int c = c0; c < c0 + 128; ++c) {
      float e = __expf(((float)Sh[r][c] - mx) * 0.125f);
      Sh[r][c] = (_Float16)e;
      sum += e;
    }
    __syncthreads();
    red[r][slot] = sum;
    __syncthreads();
    sum = 0.f;
#pragma unroll
    for (int i = 0; i < 8; ++i) sum += red[r][i];
    const float rinv = 1.f / sum;
    for (int c = c0; c < c0 + 128; ++c)
      Sh[r][c] = (_Float16)((float)Sh[r][c] * rinv);
  }
  __syncthreads();

  // O = P V : 2 m-tiles x 4 d-tiles, one 16x16 tile per wave, K = 1024
  {
    const int mt = w >> 2;
    const int nt = w & 3;
    v8f c = zv;
    const int arow = mt * 16 + (lane & 15);
    const int hi   = lane >> 4;
    for (int k0 = 0; k0 < HW; k0 += 32) {
      // A from LDS (f16, contiguous): two ds_load_b128 per lane
      const _Float16* prow = &Sh[arow][k0];
      const v8h q0 = *(const v8h*)(prow + hi * 16);
      const v8h q1 = *(const v8h*)(prow + hi * 16 + 8);
      v16h a;
#pragma unroll
      for (int j = 0; j < 8; ++j) { a[j] = q0[j]; a[8 + j] = q1[j]; }
      // B from Vt [d][s]: contiguous in s -> two global_load_b128 per lane
      v16h bf = frag_b_kc(Vt + (size_t)(nt * 16) * HW + k0, HW);
      c = wmma32(a, bf, c);
    }
    _Float16* O = ob + headOff;
#pragma unroll
    for (int i = 0; i < 8; ++i) {
      int t = t0 + mt * 16 + cm + i;
      int d = nt * 16 + cn;
      O[(size_t)t * HDIM + d] = (_Float16)c[i];
    }
  }
}

// ---------------- 5) proj GEMM + bias + residual ----------------
__global__ __launch_bounds__(256) void k_gemm_proj(const _Float16* __restrict__ wp,
                                                   const _Float16* __restrict__ ob,
                                                   const float* __restrict__ proj_b,
                                                   const float* __restrict__ x,
                                                   float* __restrict__ out) {
  const int b  = blockIdx.z;
  const int m0 = blockIdx.y * 128;
  const int n0 = blockIdx.x * 128;
  const int w  = threadIdx.x >> 5;
  const int mBase = m0 + (w >> 1) * 32;
  const int nBase = n0 + (w & 1) * 64;
  const _Float16* Ob = ob + (size_t)b * HEADS * HW * HDIM;
  const int lane = threadIdx.x & 31;

  const v8f zv = {0.f, 0.f, 0.f, 0.f, 0.f, 0.f, 0.f, 0.f};
  v8f acc[2][4];
#pragma unroll
  for (int i = 0; i < 2; ++i)
#pragma unroll
    for (int j = 0; j < 4; ++j) acc[i][j] = zv;

  for (int k0 = 0; k0 < CHAN; k0 += 32) {
    v16h a0 = frag_a_rm(wp + (size_t)mBase * CHAN + k0, CHAN);
    v16h a1 = frag_a_rm(wp + (size_t)(mBase + 16) * CHAN + k0, CHAN);
#pragma unroll
    for (int nt = 0; nt < 4; ++nt) {
      // B[k=c][n=t] gathered from O layout [h][t][d]; 16-run stays in one head
      v16h bf;
      {
        const int col   = lane & 15;
        const int kbase = k0 + (lane >> 4) * 16;
        const int hh = kbase >> 6, dd = kbase & 63;
        const _Float16* p =
            Ob + ((size_t)hh * HW + (nBase + nt * 16 + col)) * HDIM + dd;
        const v8h p0 = *(const v8h*)p;
        const v8h p1 = *(const v8h*)(p + 8);
#pragma unroll
        for (int j = 0; j < 8; ++j) { bf[j] = p0[j]; bf[8 + j] = p1[j]; }
      }
      acc[0][nt] = wmma32(a0, bf, acc[0][nt]);
      acc[1][nt] = wmma32(a1, bf, acc[1][nt]);
    }
  }

  const int cm = (lane >> 4) * 8;
  const int cn = lane & 15;
#pragma unroll
  for (int mt = 0; mt < 2; ++mt)
#pragma unroll
    for (int nt = 0; nt < 4; ++nt)
#pragma unroll
      for (int i = 0; i < 8; ++i) {
        int o = mBase + mt * 16 + cm + i;
        int t = nBase + nt * 16 + cn;
        size_t idx = ((size_t)b * CHAN + o) * HW + t;
        out[idx] = x[idx] + acc[mt][nt][i] + proj_b[o];
      }
}

extern "C" void kernel_launch(void* const* d_in, const int* in_sizes, int n_in,
                              void* d_out, int out_size, void* d_ws, size_t ws_size,
                              hipStream_t stream) {
  const float* x      = (const float*)d_in[0];
  const float* gamma  = (const float*)d_in[1];
  const float* beta   = (const float*)d_in[2];
  const float* qkv_w  = (const float*)d_in[3];
  const float* qkv_b  = (const float*)d_in[4];
  const float* proj_w = (const float*)d_in[5];
  const float* proj_b = (const float*)d_in[6];
  float* out = (float*)d_out;

  // workspace carve-up (f16 elements); total ~86 MB
  _Float16* wq  = (_Float16*)d_ws;
  _Float16* wp  = wq  + (size_t)OQKV * CHAN;
  _Float16* xnT = wp  + (size_t)CHAN * CHAN;
  _Float16* qb  = xnT + (size_t)BATCH * HW * CHAN;
  _Float16* kb  = qb  + (size_t)BATCH * HEADS * HW * HDIM;
  _Float16* vb  = kb  + (size_t)BATCH * HEADS * HW * HDIM;
  _Float16* ob  = vb  + (size_t)BATCH * HEADS * HW * HDIM;

  k_wcvt<<<(OQKV * CHAN + 255) / 256, 256, 0, stream>>>(qkv_w, proj_w, wq, wp);
  k_gn<<<BATCH * GROUPS, 256, 0, stream>>>(x, gamma, beta, xnT);
  dim3 gq(HW / 128, OQKV / 128, BATCH);
  k_gemm_qkv<<<gq, 256, 0, stream>>>(wq, xnT, qkv_b, qb, kb, vb);
  k_attn<<<BATCH * HEADS * (HW / TT), 256, 0, stream>>>(qb, kb, vb, ob);
  dim3 gp(HW / 128, CHAN / 128, BATCH);
  k_gemm_proj<<<gp, 256, 0, stream>>>(wp, ob, proj_b, x, out);
}
